// Proposal1Model_25391846654128
// MI455X (gfx1250) — compile-verified, matrix operand order checked by reading, not run on hardware
//
#include <hip/hip_runtime.h>
#include <math.h>

#define Hd   64
#define Gg   192            // 3*H
#define Tt   128
#define Bsz  4096
#define Ssz  4096
#define Edim 32

typedef __attribute__((ext_vector_type(16))) _Float16 v16h;
typedef __attribute__((ext_vector_type(8)))  float    v8f;

// ---- WMMA fragment loader: 16x32 f16 tile, row-major source p[m*ld + k] ----
// A-matrix layout (ISA 7.12.2): lanes 0-15 -> M=0..15 / K-octet 0, lanes 16-31 -> K-octet 1.
// B (KxN) is supplied via its transpose: same loader with m := n (B[k][n] = W[n][k]).
__device__ __forceinline__ v16h frag16x32(const _Float16* p, int ld) {
  int lane = threadIdx.x & 31;
  int m = lane & 15, g = lane >> 4;
  v16h f;
#pragma unroll
  for (int v = 0; v < 8; ++v) {
    int k = ((v & 4) << 2) + g * 8 + ((v & 3) << 1);
    f[2 * v]     = p[m * ld + k];
    f[2 * v + 1] = p[m * ld + k + 1];
  }
  return f;
}

// TRANS32-only activations (v_exp_f32 + v_rcp_f32, no IEEE divide chain)
__device__ __forceinline__ float fast_sigmoid(float x) {
  return __builtin_amdgcn_rcpf(1.0f + __expf(-x));
}
__device__ __forceinline__ float fast_tanh(float x) {
  return 2.0f * __builtin_amdgcn_rcpf(1.0f + __expf(-2.0f * x)) - 1.0f;
}

// ---------------------------------------------------------------------------
// Kernel 0: prep — f32 emb -> f16 emb, row norms
// ---------------------------------------------------------------------------
__global__ __launch_bounds__(256) void prep_kernel(
    const float* __restrict__ emb1, const float* __restrict__ emb2,
    _Float16* __restrict__ e1h, _Float16* __restrict__ e2h,
    float* __restrict__ n1, float* __restrict__ n2) {
  int i = blockIdx.x * 256 + threadIdx.x;
  if (i >= Ssz) return;
  float s1 = 0.f, s2 = 0.f;
#pragma unroll 4
  for (int k = 0; k < Edim; ++k) {
    float a = emb1[i * Edim + k]; e1h[i * Edim + k] = (_Float16)a; s1 += a * a;
    float b = emb2[i * Edim + k]; e2h[i * Edim + k] = (_Float16)b; s2 += b * b;
  }
  n1[i] = s1; n2[i] = s2;
}

// ---------------------------------------------------------------------------
// Kernel 1: fused 2-layer GRU scan. grid = (B/32, 2 sides), 256 thr = 8 waves.
// Weight B-fragments hoisted to registers (time-invariant); h-state A-frags
// stream from LDS each step.
// ---------------------------------------------------------------------------
__global__ __launch_bounds__(256) void gru_kernel(
    const float* __restrict__ xl, const float* __restrict__ xr,
    const float* wih0L, const float* whh0L, const float* bih0L, const float* bhh0L,
    const float* wih1L, const float* whh1L, const float* bih1L, const float* bhh1L,
    const float* wih0R, const float* whh0R, const float* bih0R, const float* bhh0R,
    const float* wih1R, const float* whh1R, const float* bih1R, const float* bhh1R,
    float* __restrict__ outl, float* __restrict__ outr) {
  __shared__ _Float16 wB[3][Gg * Hd];          // Whh0, Wih1, Whh1 (f16)
  __shared__ float bi0[Gg], bh0[Gg], bi1[Gg], bh1[Gg], w0c[Gg];
  __shared__ float gi_s[32][Gg];
  __shared__ float gh_s[32][Gg];
  __shared__ float h1f[32][Hd], h2f[32][Hd];
  __shared__ _Float16 h1h[32][Hd], h2h[32][Hd];

  int tid = threadIdx.x, wave = tid >> 5, lane = tid & 31;
  int side = blockIdx.y;
  const float* x    = side ? xr    : xl;
  const float* Wih0 = side ? wih0R : wih0L;
  const float* Whh0 = side ? whh0R : whh0L;
  const float* Bih0 = side ? bih0R : bih0L;
  const float* Bhh0 = side ? bhh0R : bhh0L;
  const float* Wih1 = side ? wih1R : wih1L;
  const float* Whh1 = side ? whh1R : whh1L;
  const float* Bih1 = side ? bih1R : bih1L;
  const float* Bhh1 = side ? bhh1R : bhh1L;
  float* outp = side ? outr : outl;
  int r0 = blockIdx.x * 32;

  for (int i = tid; i < Gg * Hd; i += 256) {
    wB[0][i] = (_Float16)Whh0[i];
    wB[1][i] = (_Float16)Wih1[i];
    wB[2][i] = (_Float16)Whh1[i];
  }
  for (int i = tid; i < Gg; i += 256) {
    bi0[i] = Bih0[i]; bh0[i] = Bhh0[i];
    bi1[i] = Bih1[i]; bh1[i] = Bhh1[i];
    w0c[i] = Wih0[i];                              // Wih layer0 is [192,1]
  }
  for (int i = tid; i < 32 * Hd; i += 256) {
    int b = i >> 6, j = i & 63;
    h1f[b][j] = 0.f; h2f[b][j] = 0.f;
    h1h[b][j] = (_Float16)0.f; h2h[b][j] = (_Float16)0.f;
  }
  __syncthreads();

  // ---- hoist time-invariant weight fragments + biases into registers ----
  int col = lane & 15, rowoff = (lane >> 4) << 3;
  v16h wb0f[3][2]; float bv0[3]; int mt0[3], nt0[3];
#pragma unroll
  for (int q = 0; q < 3; ++q) {
    int tt = wave + q * 8;
    int mt = tt / 12, nt = tt - mt * 12;
    mt0[q] = mt; nt0[q] = nt;
    wb0f[q][0] = frag16x32(&wB[0][nt * 16 * Hd], Hd);
    wb0f[q][1] = frag16x32(&wB[0][nt * 16 * Hd] + 32, Hd);
    bv0[q] = bh0[nt * 16 + col];
  }
  v16h wb1f[6][2]; float bv1[6]; int mt1[6], nt1[6], wh1[6];
#pragma unroll
  for (int q = 0; q < 6; ++q) {
    int tt = wave + q * 8;
    int which = tt / 24, t2 = tt - which * 24;
    int mt = t2 / 12, nt = t2 - mt * 12;
    mt1[q] = mt; nt1[q] = nt; wh1[q] = which;
    const _Float16* Bp = which ? &wB[2][nt * 16 * Hd] : &wB[1][nt * 16 * Hd];
    wb1f[q][0] = frag16x32(Bp, Hd);
    wb1f[q][1] = frag16x32(Bp + 32, Hd);
    bv1[q] = which ? bh1[nt * 16 + col] : bi1[nt * 16 + col];
  }

  for (int t = 0; t < Tt; ++t) {
    // ---- layer0: gi0 elementwise (input dim 1), gh0 = h1 @ Whh0^T + bhh0 ----
    for (int i = tid; i < 32 * Gg; i += 256) {
      int b = i / Gg, j = i - b * Gg;
      gi_s[b][j] = x[(r0 + b) * Tt + t] * w0c[j] + bi0[j];
    }
#pragma unroll
    for (int q = 0; q < 3; ++q) {
      int mt = mt0[q], nt = nt0[q];
      v8f c;
#pragma unroll
      for (int v = 0; v < 8; ++v) c[v] = bv0[q];
      c = __builtin_amdgcn_wmma_f32_16x16x32_f16(false, frag16x32(&h1h[mt * 16][0], Hd),
            false, wb0f[q][0], (short)0, c, false, false);
      c = __builtin_amdgcn_wmma_f32_16x16x32_f16(false, frag16x32(&h1h[mt * 16][0] + 32, Hd),
            false, wb0f[q][1], (short)0, c, false, false);
      int cc = nt * 16 + col, rb = mt * 16 + rowoff;
#pragma unroll
      for (int v = 0; v < 8; ++v) gh_s[rb + v][cc] = c[v];
    }
    __syncthreads();
    for (int i = tid; i < 32 * Hd; i += 256) {
      int b = i >> 6, j = i & 63;
      float r = fast_sigmoid(gi_s[b][j] + gh_s[b][j]);
      float z = fast_sigmoid(gi_s[b][j + 64] + gh_s[b][j + 64]);
      float n = fast_tanh(gi_s[b][j + 128] + r * gh_s[b][j + 128]);
      float hn = (1.f - z) * n + z * h1f[b][j];
      h1f[b][j] = hn; h1h[b][j] = (_Float16)hn;
    }
    __syncthreads();
    // ---- layer1: gi1 = h1 @ Wih1^T + bih1 ; gh1 = h2 @ Whh1^T + bhh1 ----
#pragma unroll
    for (int q = 0; q < 6; ++q) {
      int mt = mt1[q], nt = nt1[q], which = wh1[q];
      const _Float16* A = which ? &h2h[mt * 16][0] : &h1h[mt * 16][0];
      v8f c;
#pragma unroll
      for (int v = 0; v < 8; ++v) c[v] = bv1[q];
      c = __builtin_amdgcn_wmma_f32_16x16x32_f16(false, frag16x32(A, Hd),
            false, wb1f[q][0], (short)0, c, false, false);
      c = __builtin_amdgcn_wmma_f32_16x16x32_f16(false, frag16x32(A + 32, Hd),
            false, wb1f[q][1], (short)0, c, false, false);
      int cc = nt * 16 + col, rb = mt * 16 + rowoff;
      if (which) {
#pragma unroll
        for (int v = 0; v < 8; ++v) gh_s[rb + v][cc] = c[v];
      } else {
#pragma unroll
        for (int v = 0; v < 8; ++v) gi_s[rb + v][cc] = c[v];
      }
    }
    __syncthreads();
    for (int i = tid; i < 32 * Hd; i += 256) {
      int b = i >> 6, j = i & 63;
      float r = fast_sigmoid(gi_s[b][j] + gh_s[b][j]);
      float z = fast_sigmoid(gi_s[b][j + 64] + gh_s[b][j + 64]);
      float n = fast_tanh(gi_s[b][j + 128] + r * gh_s[b][j + 128]);
      float hn = (1.f - z) * n + z * h2f[b][j];
      h2f[b][j] = hn; h2h[b][j] = (_Float16)hn;
    }
    __syncthreads();
  }
  for (int i = tid; i < 32 * Hd; i += 256) {
    int b = i >> 6, j = i & 63;
    outp[(r0 + b) * Hd + j] = h2f[b][j];
  }
}

// ---------------------------------------------------------------------------
// Kernel 2: cdist (WMMA) + exp(-d) + top-21 selection + side features.
// y_ctx tiles double-buffered into LDS with global_load_async_to_lds_b128,
// overlapped with WMMA distance compute; synced with s_wait_asynccnt.
// grid = B/16 blocks, 256 thr = 8 waves; 16 rows/block, 16 sel-threads/row.
// ---------------------------------------------------------------------------
__global__ __launch_bounds__(256) void side_kernel(
    const _Float16* __restrict__ embh, const float* __restrict__ norms,
    const int* __restrict__ idx, const float* __restrict__ yctx,
    float* __restrict__ feats, int colOff) {
  __shared__ _Float16 aT[16 * Edim];
  __shared__ float na[16];
  __shared__ float wbuf[16][512];
  __shared__ float ybuf[2][16][512];
  __shared__ float candw[16][16 * 21];
  __shared__ float candy[16][16 * 21];

  int tid = threadIdx.x, wave = tid >> 5, lane = tid & 31;
  int r0 = blockIdx.x * 16;
  unsigned long long ybase = (unsigned long long)(uintptr_t)yctx;

  // async-stage one 16x512 f32 y-tile: 2048 x 16B, 8 per thread
  auto issue_y = [&](int ch, int buf) {
    unsigned lbase = (unsigned)(uintptr_t)&ybuf[buf][0][0];
#pragma unroll
    for (int j = 0; j < 8; ++j) {
      unsigned q    = (unsigned)tid + (unsigned)j * 256u;   // 0..2047
      unsigned fob  = q << 4;                               // byte off in tile
      unsigned row  = fob >> 11;                            // 2048 B per row-chunk
      unsigned colb = fob & 2047u;
      unsigned goff = ((unsigned)(r0 + (int)row) << 14) + ((unsigned)ch << 11) + colb;
      unsigned ldsa = lbase + fob;
      asm volatile("global_load_async_to_lds_b128 %0, %1, %2"
                   :: "v"(ldsa), "v"(goff), "s"(ybase) : "memory");
    }
  };

  issue_y(0, 0);

  for (int i = tid; i < 16 * Edim; i += 256) {
    int m = i >> 5, k = i & 31;
    aT[m * Edim + k] = embh[idx[r0 + m] * Edim + k];
  }
  if (tid < 16) na[tid] = norms[idx[r0 + tid]];

  float tw[21], ty[21];
#pragma unroll
  for (int q = 0; q < 21; ++q) { tw[q] = -1.f; ty[q] = 0.f; }
  float wmin = -1.f; int imin = 0;
  __syncthreads();

  int brow = tid >> 4, tc = tid & 15;
  for (int ch = 0; ch < 8; ++ch) {
    int c0 = ch * 512;
    if (ch < 7) {
      issue_y(ch + 1, (ch + 1) & 1);
      // current chunk's 8 loads complete (per-thread in-order); next 8 stay in flight
      asm volatile("s_wait_asynccnt 8" ::: "memory");
    } else {
      asm volatile("s_wait_asynccnt 0" ::: "memory");
    }
    for (int nt = wave; nt < 32; nt += 8) {
      int cb = c0 + nt * 16;
      v8f c = {};
      c = __builtin_amdgcn_wmma_f32_16x16x32_f16(false, frag16x32(aT, Edim),
            false, frag16x32(&embh[cb * Edim], Edim), (short)0, c, false, false);
      int n = lane & 15, rb = (lane >> 4) << 3;
      float ns = norms[cb + n];
#pragma unroll
      for (int v = 0; v < 8; ++v) {
        float d2 = na[rb + v] + ns - 2.f * c[v];
        d2 = d2 > 0.f ? d2 : 0.f;
        wbuf[rb + v][nt * 16 + n] = __expf(-__fsqrt_rn(d2));   // TAU = 1
      }
    }
    __syncthreads();   // wbuf ready; every thread's async wait retired
    for (int jj = 0; jj < 32; ++jj) {
      int s = tc + jj * 16;
      float w = wbuf[brow][s];
      if (w > wmin) {
        float yv = ybuf[ch & 1][brow][s];
        tw[imin] = w; ty[imin] = yv;
        wmin = tw[0]; imin = 0;
#pragma unroll
        for (int q = 1; q < 21; ++q) if (tw[q] < wmin) { wmin = tw[q]; imin = q; }
      }
    }
    __syncthreads();   // ybuf[ch&1] free before chunk ch+2 refills it
  }
#pragma unroll
  for (int q = 0; q < 21; ++q) {
    candw[brow][tc * 21 + q] = tw[q];
    candy[brow][tc * 21 + q] = ty[q];
  }
  __syncthreads();
  if (tid < 16) {
    int b = tid;
    float wk[20], yk[20];
    for (int round = 0; round < 21; ++round) {       // round 0 = self-match, dropped
      float best = -2.f; int bi = 0;
      for (int q = 0; q < 16 * 21; ++q) {
        float v = candw[b][q];
        if (v > best) { best = v; bi = q; }
      }
      candw[b][bi] = -2.f;
      if (round > 0) { wk[round - 1] = best; yk[round - 1] = candy[b][bi]; }
    }
    float ws = 0.f, wy = 0.f, sy = 0.f;
#pragma unroll
    for (int q = 0; q < 20; ++q) { ws += wk[q]; wy += wk[q] * yk[q]; sy += yk[q]; }
    float my = sy * (1.f / 20.f), v2 = 0.f;
#pragma unroll
    for (int q = 0; q < 20; ++q) { float d = yk[q] - my; v2 += d * d; }
    float* fp = &feats[(r0 + b) * 8 + colOff];
    fp[0] = wy * __builtin_amdgcn_rcpf(ws);
    fp[1] = ws;
    fp[2] = __fsqrt_rn(v2 * (1.f / 19.f));           // ddof=1
  }
}

// ---------------------------------------------------------------------------
// Kernel 3: mean_ts / std_ts heads + err1 per-row terms
// ---------------------------------------------------------------------------
__global__ __launch_bounds__(256) void ts_kernel(
    const float* __restrict__ outl, const float* __restrict__ outr,
    const float* mW, const float* mb, const float* sW, const float* sb,
    const float* __restrict__ y, float* __restrict__ feats, float* __restrict__ err1t) {
  int b = blockIdx.x * 256 + threadIdx.x;
  if (b >= Bsz) return;
  float m = mb[0], s = sb[0];
  for (int j = 0; j < Hd; ++j) {
    float l = outl[b * Hd + j], r = outr[b * Hd + j];
    m += l * mW[j] + r * mW[Hd + j];
    s += l * sW[j] + r * sW[Hd + j];
  }
  feats[b * 8 + 6] = m;
  feats[b * 8 + 7] = s;
  float d = y[b] - m;
  err1t[b] = d * d * __expf(-s) + s;
}

// ---------------------------------------------------------------------------
// Kernel 4: 8 -> 64 (ReLU) -> mean/std heads, mean_o out, err2 terms
// ---------------------------------------------------------------------------
__global__ __launch_bounds__(256) void head_kernel(
    const float* __restrict__ feats, const float* W1, const float* b1,
    const float* mW, const float* mb, const float* sW, const float* sb,
    const float* __restrict__ y, float* __restrict__ out, float* __restrict__ err2t) {
  int b = blockIdx.x * 256 + threadIdx.x;
  if (b >= Bsz) return;
  float f[8];
#pragma unroll
  for (int i = 0; i < 8; ++i) f[i] = feats[b * 8 + i];
  float mo = mb[0], so = sb[0];
  for (int k = 0; k < 64; ++k) {
    float a = b1[k];
#pragma unroll
    for (int i = 0; i < 8; ++i) a += W1[k * 8 + i] * f[i];
    a = a > 0.f ? a : 0.f;
    mo += mW[k] * a; so += sW[k] * a;
  }
  out[2 + b] = mo;
  float d = y[b] - mo;
  err2t[b] = d * d * __expf(-so) + so;
}

// ---------------------------------------------------------------------------
// Kernel 5: deterministic tree reduction for err1/err2 means
// ---------------------------------------------------------------------------
__global__ __launch_bounds__(256) void reduce_kernel(
    const float* __restrict__ e1, const float* __restrict__ e2, float* __restrict__ out) {
  __shared__ float s1[256], s2[256];
  int tid = threadIdx.x;
  float a = 0.f, b = 0.f;
  for (int i = tid; i < Bsz; i += 256) { a += e1[i]; b += e2[i]; }
  s1[tid] = a; s2[tid] = b;
  __syncthreads();
  for (int st = 128; st > 0; st >>= 1) {
    if (tid < st) { s1[tid] += s1[tid + st]; s2[tid] += s2[tid + st]; }
    __syncthreads();
  }
  if (tid == 0) { out[0] = s1[0] / (float)Bsz; out[1] = s2[0] / (float)Bsz; }
}

// ---------------------------------------------------------------------------
extern "C" void kernel_launch(void* const* d_in, const int* in_sizes, int n_in,
                              void* d_out, int out_size, void* d_ws, size_t ws_size,
                              hipStream_t stream) {
  const float* xl   = (const float*)d_in[0];
  const float* xr   = (const float*)d_in[1];
  const float* y    = (const float*)d_in[2];
  const float* y1   = (const float*)d_in[3];
  const float* y2   = (const float*)d_in[4];
  const int*   idx1 = (const int*)d_in[5];
  const int*   idx2 = (const int*)d_in[6];
  // params (insertion order): gl[2 layers x (Wih,Whh,bih,bhh)], gr[...], heads, embs, MLP
  const float* gl[8]; const float* gr[8];
  for (int i = 0; i < 8; ++i) { gl[i] = (const float*)d_in[7 + i]; gr[i] = (const float*)d_in[15 + i]; }
  const float* meanW = (const float*)d_in[23];
  const float* meanB = (const float*)d_in[24];
  const float* stdW  = (const float*)d_in[25];
  const float* stdB  = (const float*)d_in[26];
  const float* emb1  = (const float*)d_in[27];
  const float* emb2  = (const float*)d_in[28];
  const float* ol1W  = (const float*)d_in[29];
  const float* ol1b  = (const float*)d_in[30];
  const float* omW   = (const float*)d_in[31];
  const float* omb   = (const float*)d_in[32];
  const float* osW   = (const float*)d_in[33];
  const float* osb   = (const float*)d_in[34];

  char* ws = (char*)d_ws;
  size_t o = 0;
  float* out_l = (float*)(ws + o); o += (size_t)Bsz * Hd * 4;
  float* out_r = (float*)(ws + o); o += (size_t)Bsz * Hd * 4;
  float* feats = (float*)(ws + o); o += (size_t)Bsz * 8 * 4;
  float* err1t = (float*)(ws + o); o += (size_t)Bsz * 4;
  float* err2t = (float*)(ws + o); o += (size_t)Bsz * 4;
  float* n1    = (float*)(ws + o); o += (size_t)Ssz * 4;
  float* n2    = (float*)(ws + o); o += (size_t)Ssz * 4;
  _Float16* e1h = (_Float16*)(ws + o); o += (size_t)Ssz * Edim * 2;
  _Float16* e2h = (_Float16*)(ws + o); o += (size_t)Ssz * Edim * 2;

  float* out = (float*)d_out;

  prep_kernel<<<Ssz / 256, 256, 0, stream>>>(emb1, emb2, e1h, e2h, n1, n2);

  gru_kernel<<<dim3(Bsz / 32, 2), 256, 0, stream>>>(
      xl, xr,
      gl[0], gl[1], gl[2], gl[3], gl[4], gl[5], gl[6], gl[7],
      gr[0], gr[1], gr[2], gr[3], gr[4], gr[5], gr[6], gr[7],
      out_l, out_r);

  side_kernel<<<Bsz / 16, 256, 0, stream>>>(e1h, n1, idx1, y1, feats, 0);
  side_kernel<<<Bsz / 16, 256, 0, stream>>>(e2h, n2, idx2, y2, feats, 3);

  ts_kernel<<<Bsz / 256, 256, 0, stream>>>(out_l, out_r, meanW, meanB, stdW, stdB,
                                           y, feats, err1t);
  head_kernel<<<Bsz / 256, 256, 0, stream>>>(feats, ol1W, ol1b, omW, omb, osW, osb,
                                             y, out, err2t);
  reduce_kernel<<<1, 256, 0, stream>>>(err1t, err2t, out);
}
